// MDAM_51513837748610
// MI455X (gfx1250) — matrix operand order repaired and checked by reading, not compiled, not run on hardware
//
#include <hip/hip_runtime.h>
#include <math.h>

#define EPS_ 1e-5f

typedef __attribute__((ext_vector_type(16))) __bf16 bf16x16;
typedef __attribute__((ext_vector_type(8)))  float  f32x8;

__device__ __forceinline__ float sigmoidf_(float x){ return 1.f/(1.f+expf(-x)); }

// ---- WMMA helper: A and B rows are two 8-bf16 runs (at +0 and +16) ----
__device__ __forceinline__ f32x8 wmma_bf16(const __bf16* arow0, const __bf16* arow1,
                                           const __bf16* brow, f32x8 acc) {
  bf16x16 a, b;
  #pragma unroll
  for (int e = 0; e < 8; ++e) { a[e] = arow0[e]; a[e + 8] = arow1[e]; }
  #pragma unroll
  for (int e = 0; e < 8; ++e) { b[e] = brow[e]; b[e + 8] = brow[e + 16]; }
  return __builtin_amdgcn_wmma_f32_16x16x32_bf16(false, a, false, b, (short)0, acc, false, false);
}

// ================= K0: weight prep (fold BN scales, bf16 convert, pw channel-shuffle perm) =================
__global__ __launch_bounds__(256)
void k_wprep(const float* __restrict__ w1, const float* __restrict__ g1,
             const float* __restrict__ pw, const float* __restrict__ gpw,
             const float* __restrict__ d1w, const float* __restrict__ d1g, const float* __restrict__ d1b,
             const float* __restrict__ d3w, const float* __restrict__ d3g, const float* __restrict__ d3b,
             const float* __restrict__ d5w, const float* __restrict__ d5g, const float* __restrict__ d5b,
             __bf16* __restrict__ w1bf, __bf16* __restrict__ pwp,
             float* __restrict__ dwf, float* __restrict__ dbias)
{
  const float bs = rsqrtf(1.f + EPS_);
  int i = blockIdx.x * 256 + threadIdx.x;
  if (i < 768 * 128) {
    int o = i / 128;
    w1bf[i] = (__bf16)(w1[i] * g1[o] * bs);
    int o2 = i / 768, c = i % 768;
    int n = (c % 24) * 32 + (c / 24);          // channel_shuffle(32) folded into pw indexing
    pwp[i] = (__bf16)(pw[o2 * 768 + n] * gpw[o2] * bs);
  }
  if (i < 768) {
    float s1 = d1g[i] * bs, s3 = d3g[i] * bs, s5 = d5g[i] * bs;
    dwf[i * 35 + 0] = d1w[i] * s1;
    for (int t = 0; t < 9;  ++t) dwf[i * 35 + 1  + t] = d3w[i * 9  + t] * s3;
    for (int t = 0; t < 25; ++t) dwf[i * 35 + 10 + t] = d5w[i * 25 + t] * s5;
    dbias[i * 3 + 0] = d1b[i]; dbias[i * 3 + 1] = d3b[i]; dbias[i * 3 + 2] = d5b[i];
  }
}

// ================= K1: axis means =================
__global__ __launch_bounds__(256)
void k_means(const float* __restrict__ x, float* __restrict__ hm, float* __restrict__ wm)
{
  int bc = blockIdx.x;                       // b*128+c, 1024 blocks
  const float* p = x + (size_t)bc * 19600;
  int tid = threadIdx.x;
  if (tid < 140) {
    float cs = 0.f;
    for (int h = 0; h < 140; ++h) cs += p[h * 140 + tid];
    wm[bc * 140 + tid] = cs * (1.f / 140.f);
    float rs = 0.f;
    for (int w = 0; w < 140; ++w) rs += p[tid * 140 + w];
    hm[bc * 140 + tid] = rs * (1.f / 140.f);
  }
}

// ================= K2: DSAB gating (dwconv1d x2 + swish + groupnorm + sigmoid) =================
__global__ __launch_bounds__(256)
void k_dsab(const float* __restrict__ hm, const float* __restrict__ wm,
            const float* __restrict__ w3, const float* __restrict__ b3,
            const float* __restrict__ w5, const float* __restrict__ b5,
            const float* __restrict__ w7, const float* __restrict__ b7,
            const float* __restrict__ w9, const float* __restrict__ b9,
            const float* __restrict__ hg, const float* __restrict__ hb,
            const float* __restrict__ wg, const float* __restrict__ wb,
            float* __restrict__ hx, float* __restrict__ wx)
{
  __shared__ float vbuf[32][140];
  __shared__ float t1[32][140];
  __shared__ float t2[32][140];
  __shared__ float red[2][256];
  int id = blockIdx.x;                       // 64 = b(8) x q(4) x axis(2)
  int axis = id & 1;
  int q = (id >> 1) & 3;
  int b = id >> 3;
  const float* src   = axis ? wm : hm;
  const float* gamma = axis ? wg : hg;
  const float* beta  = axis ? wb : hb;
  float* dst = axis ? wx : hx;
  const float* wA = (q < 2) ? w3 : w7; const float* bA = (q < 2) ? b3 : b7; int kA = (q < 2) ? 3 : 7;
  const float* wB = (q < 2) ? w5 : w9; const float* bB = (q < 2) ? b5 : b9; int kB = (q < 2) ? 5 : 9;
  int tid = threadIdx.x;
  for (int i = tid; i < 32 * 140; i += 256) { int j = i / 140, l = i % 140; vbuf[j][l] = src[(b * 128 + q * 32 + j) * 140 + l]; }
  __syncthreads();
  for (int i = tid; i < 32 * 140; i += 256) {
    int j = i / 140, l = i % 140, pad = kA / 2;
    float s = bA[j];
    for (int t = 0; t < kA; ++t) { int ll = l + t - pad; if (ll >= 0 && ll < 140) s += wA[j * kA + t] * vbuf[j][ll]; }
    t1[j][l] = s * sigmoidf_(s);
  }
  __syncthreads();
  for (int i = tid; i < 32 * 140; i += 256) {
    int j = i / 140, l = i % 140, pad = kB / 2;
    float s = bB[j];
    for (int t = 0; t < kB; ++t) { int ll = l + t - pad; if (ll >= 0 && ll < 140) s += wB[j * kB + t] * t1[j][ll]; }
    t2[j][l] = s * sigmoidf_(s);
  }
  __syncthreads();
  float ps = 0.f, pq = 0.f;
  for (int i = tid; i < 32 * 140; i += 256) { float v = t2[i / 140][i % 140]; ps += v; pq += v * v; }
  red[0][tid] = ps; red[1][tid] = pq; __syncthreads();
  for (int s = 128; s > 0; s >>= 1) {
    if (tid < s) { red[0][tid] += red[0][tid + s]; red[1][tid] += red[1][tid + s]; }
    __syncthreads();
  }
  float mean = red[0][0] * (1.f / 4480.f);
  float var  = red[1][0] * (1.f / 4480.f) - mean * mean;
  float rs = rsqrtf(var + EPS_);
  for (int i = tid; i < 32 * 140; i += 256) {
    int j = i / 140, l = i % 140, c = q * 32 + j;
    float v = (t2[j][l] - mean) * rs * gamma[c] + beta[c];
    dst[(b * 128 + c) * 140 + l] = sigmoidf_(v);
  }
}

// ================= K3: apply axis gates, write gated x to out, fused 7x7 avg-pool -> y =================
__global__ __launch_bounds__(256)
void k_gate_pool(const float* __restrict__ x, const float* __restrict__ hx,
                 const float* __restrict__ wx, float* __restrict__ out,
                 float* __restrict__ y)
{
  __shared__ float tmp[140];
  int bc = blockIdx.x;
  const float* px = x + (size_t)bc * 19600;
  float* po = out + (size_t)bc * 19600;
  const float* ph = hx + bc * 140;
  const float* pw = wx + bc * 140;
  int tid = threadIdx.x;
  float wxv = (tid < 140) ? pw[tid] : 0.f;
  for (int hh = 0; hh < 20; ++hh) {
    float s = 0.f;
    for (int dh = 0; dh < 7; ++dh) {
      int h = hh * 7 + dh;
      if (tid < 140) {
        float v = px[h * 140 + tid] * ph[h] * wxv;
        po[h * 140 + tid] = v;
        s += v;
      }
    }
    if (tid < 140) tmp[tid] = s;
    __syncthreads();
    if (tid < 20) {
      float t = 0.f;
      for (int dw = 0; dw < 7; ++dw) t += tmp[tid * 7 + dw];
      y[bc * 400 + hh * 20 + tid] = t * (1.f / 49.f);
    }
    __syncthreads();
  }
}

// ================= K4: per-batch groupnorm(groups=1) on pooled y =================
__global__ __launch_bounds__(256)
void k_ynorm(const float* __restrict__ y, const float* __restrict__ g,
             const float* __restrict__ bta, float* __restrict__ yn)
{
  __shared__ float red[2][256];
  int b = blockIdx.x, tid = threadIdx.x;
  const float* p = y + b * 51200;
  float ps = 0.f, pq = 0.f;
  for (int i = tid; i < 51200; i += 256) { float v = p[i]; ps += v; pq += v * v; }
  red[0][tid] = ps; red[1][tid] = pq; __syncthreads();
  for (int s = 128; s > 0; s >>= 1) {
    if (tid < s) { red[0][tid] += red[0][tid + s]; red[1][tid] += red[1][tid + s]; }
    __syncthreads();
  }
  float mean = red[0][0] / 51200.f;
  float var  = red[1][0] / 51200.f - mean * mean;
  float rs = rsqrtf(var + EPS_);
  for (int i = tid; i < 51200; i += 256) {
    int c = i / 400;
    yn[b * 51200 + i] = (p[i] - mean) * rs * g[c] + bta[c];
  }
}

// ================= K5: channel attention head, WMMA q.k^T + softmax + gate partial =================
// Single launch: each block decodes (b, head, dir) and selects the h- or w-axis weight set itself.
__global__ __launch_bounds__(32)
void k_attn(const float* __restrict__ yn,
            const float* __restrict__ qhw, const float* __restrict__ khw, const float* __restrict__ vhw,
            const float* __restrict__ qww, const float* __restrict__ kww, const float* __restrict__ vww,
            float* __restrict__ gbuf)
{
  __shared__ float  yl[16][400];
  __shared__ __bf16 qs[16][420];
  __shared__ __bf16 ks[16][420];
  __shared__ float  vml[16];
  int id = blockIdx.x;                       // 128 = b(8) x hd(8) x dir(2)
  int b = id >> 4;
  int hd = (id >> 1) & 7;
  int dir = id & 1;
  const float* qw = dir ? qww : qhw;
  const float* kw = dir ? kww : khw;
  const float* vw = dir ? vww : vhw;
  int tid = threadIdx.x;
  for (int i = tid; i < 16 * 400; i += 32) { int d = i / 400, p = i % 400; yl[d][p] = yn[(b * 128 + hd * 16 + d) * 400 + p]; }
  if (tid < 16) vml[tid] = 0.f;
  __syncthreads();
  auto conv = [&](int d, int p, const float* wt, int K) -> float {
    int hh = p / 20, ww = p % 20, pad = K / 2;
    float s = 0.f;
    for (int t = 0; t < K; ++t) {
      int q_ = (dir == 0) ? hh + t - pad : ww + t - pad;
      if (q_ < 0 || q_ >= 20) continue;
      int pp = (dir == 0) ? q_ * 20 + ww : hh * 20 + q_;
      s += wt[(hd * 16 + d) * K + t] * yl[d][pp];
    }
    return s;
  };
  for (int i = tid; i < 16 * 416; i += 32) {
    int d = i / 416, p = i % 416;
    qs[d][p] = (p < 400) ? (__bf16)conv(d, p, qw, 7)  : (__bf16)0.f;
    ks[d][p] = (p < 400) ? (__bf16)conv(d, p, kw, 11) : (__bf16)0.f;
  }
  for (int i = tid; i < 6400; i += 32) {
    int d = i / 400, p = i % 400;
    atomicAdd(&vml[d], conv(d, p, vw, 21) * (1.f / 400.f));
  }
  __syncthreads();
  int m = tid & 15, g = tid >> 4;
  f32x8 acc = {};
  for (int kc = 0; kc < 13; ++kc) {
    int k0 = kc * 32;
    acc = wmma_bf16(&qs[m][k0 + g * 8], &qs[m][k0 + 16 + g * 8], &ks[m][k0 + g * 8], acc);
  }
  #pragma unroll
  for (int r = 0; r < 8; ++r) {
    float s = acc[r] * 0.25f;                               // HD^-0.5
    float mx = s;
    for (int msk = 1; msk < 16; msk <<= 1) mx = fmaxf(mx, __shfl_xor(mx, msk, 32));
    float e = expf(s - mx);
    float sm = e;
    for (int msk = 1; msk < 16; msk <<= 1) sm += __shfl_xor(sm, msk, 32);
    float a = e / sm;
    float p = a * vml[m];                                   // mean_n(a.v) == sum_e a*vmean
    for (int msk = 1; msk < 16; msk <<= 1) p += __shfl_xor(p, msk, 32);
    if (m == 0) gbuf[dir * 1024 + b * 128 + hd * 16 + r + 8 * g] = p;
  }
}

// ================= K6: apply channel gate to out, emit bf16 copy for the MCB GEMMs =================
__global__ __launch_bounds__(256)
void k_chgate(float* __restrict__ out, const float* __restrict__ gbuf,
              __bf16* __restrict__ xbf)
{
  int i = blockIdx.x * 256 + threadIdx.x;
  if (i >= 8 * 128 * 19600) return;
  int c = (i / 19600) & 127;
  int b = i / (128 * 19600);
  float gt = sigmoidf_(gbuf[b * 128 + c] + gbuf[1024 + b * 128 + c]);
  float v = out[i] * gt;
  out[i] = v;
  xbf[i] = (__bf16)v;
}

// ================= K7: fused MCB (1x1 WMMA GEMM -> dwconvs -> shuffled 1x1 WMMA GEMM + residual) =====
__global__ __launch_bounds__(256)
void k_mcb(const __bf16* __restrict__ xbf, const __bf16* __restrict__ w1bf,
           const __bf16* __restrict__ pwp, const float* __restrict__ dwf,
           const float* __restrict__ dbias, const float* __restrict__ b1,
           const float* __restrict__ pwb, float* __restrict__ out)
{
  __shared__ __bf16 xs[160][136];     // halo px (12x12 pad->160) x ic 128 (pad 136)
  __shared__ __bf16 wlds[64][132];    // W1 chunk: oc64 x ic128 (pad 132)
  __shared__ __bf16 c1s[160][72];     // halo px x oc64 (pad 72)
  __shared__ __bf16 mcs[64][72];      // inner px 64 x oc64 (pad 72)
  __shared__ __bf16 pwlds[128][68];   // pw chunk: oc128 x k64 (pad 68)

  const int tid = threadIdx.x;
  const int lane = tid & 31;
  const int wv = tid >> 5;                    // 0..7
  const int m = lane & 15, g = lane >> 4;
  const int b = blockIdx.z;
  const int th = blockIdx.y, tw = blockIdx.x; // 18x18 tiles of 8x8
  const int h0 = th * 8 - 2, w0 = tw * 8 - 2;

  // load x tile+halo (zero-padded), semi-coalesced along w
  for (int i = tid; i < 160 * 128; i += 256) {
    int px = i % 160, ic = i / 160;
    __bf16 v = (__bf16)0.f;
    if (px < 144) {
      int py = px / 12, pxx = px % 12;
      int h = h0 + py, w = w0 + pxx;
      if (h >= 0 && h < 140 && w >= 0 && w < 140)
        v = xbf[((b * 128 + ic) * 140 + h) * 140 + w];
    }
    xs[px][ic] = v;
  }

  f32x8 acc2[4] = {};                         // persistent pw accumulators (oc tile wv, 4 N tiles)

  for (int ch = 0; ch < 12; ++ch) {
    __syncthreads();                          // protect previous-iter LDS reads (and xs fill on iter 0)
    for (int i = tid; i < 64 * 128; i += 256) { int oc = i >> 7, ic = i & 127; wlds[oc][ic] = w1bf[(ch * 64 + oc) * 128 + ic]; }
    for (int i = tid; i < 128 * 64; i += 256) { int o = i >> 6, k = i & 63;   pwlds[o][k] = pwp[o * 768 + ch * 64 + k]; }
    if (ch + 1 < 12) {                        // pre-stage next chunk's weights (global_prefetch_b8)
      __builtin_prefetch(&w1bf[(ch + 1) * 64 * 128 + tid * 32]);
      __builtin_prefetch(&pwp[(tid >> 1) * 768 + (ch + 1) * 64 + (tid & 1) * 32]);
    }
    __syncthreads();

    // GEMM1: c1 = relu(W1.x + b1), M=64 oc x N=160 px x K=128
    for (int t = wv; t < 40; t += 8) {
      int mt = t & 3, nt = t >> 2;
      f32x8 acc = {};
      #pragma unroll
      for (int ks2 = 0; ks2 < 4; ++ks2) {
        int k0 = ks2 * 32;
        acc = wmma_bf16(&wlds[mt * 16 + m][k0 + g * 8], &wlds[mt * 16 + m][k0 + 16 + g * 8],
                        &xs[nt * 16 + m][k0 + g * 8], acc);
      }
      int px = nt * 16 + m;
      #pragma unroll
      for (int r = 0; r < 8; ++r) {
        int ocl = mt * 16 + r + 8 * g;
        float v = acc[r] + b1[ch * 64 + ocl];
        c1s[px][ocl] = (__bf16)(v > 0.f ? v : 0.f);
      }
    }
    __syncthreads();

    // multi-scale depthwise (1x1 + 3x3 + 5x5, each relu6) into mcs
    for (int i = tid; i < 64 * 64; i += 256) {
      int c = i >> 6, p = i & 63;
      int ph = p >> 3, pw = p & 7;
      int c6 = ch * 64 + c;
      const float* wf = &dwf[c6 * 35];
      float ctr = (float)c1s[(ph + 2) * 12 + (pw + 2)][c];
      float s1v = wf[0] * ctr;
      float s3 = 0.f;
      #pragma unroll
      for (int dy = 0; dy < 3; ++dy)
        #pragma unroll
        for (int dx = 0; dx < 3; ++dx)
          s3 += wf[1 + dy * 3 + dx] * (float)c1s[(ph + 1 + dy) * 12 + (pw + 1 + dx)][c];
      float s5 = 0.f;
      #pragma unroll
      for (int dy = 0; dy < 5; ++dy)
        #pragma unroll
        for (int dx = 0; dx < 5; ++dx)
          s5 += wf[10 + dy * 5 + dx] * (float)c1s[(ph + dy) * 12 + (pw + dx)][c];
      float r1 = fminf(fmaxf(s1v + dbias[c6 * 3 + 0], 0.f), 6.f);
      float r3 = fminf(fmaxf(s3  + dbias[c6 * 3 + 1], 0.f), 6.f);
      float r5 = fminf(fmaxf(s5  + dbias[c6 * 3 + 2], 0.f), 6.f);
      mcs[p][c] = (__bf16)(r1 + r3 + r5);
    }
    __syncthreads();

    // GEMM2: out += pw_perm_chunk . mcs, M=128 oc (wave-tiled) x N=64 px x K=64
    #pragma unroll
    for (int nt = 0; nt < 4; ++nt) {
      #pragma unroll
      for (int ks2 = 0; ks2 < 2; ++ks2) {
        int k0 = ks2 * 32;
        acc2[nt] = wmma_bf16(&pwlds[wv * 16 + m][k0 + g * 8], &pwlds[wv * 16 + m][k0 + 16 + g * 8],
                             &mcs[nt * 16 + m][k0 + g * 8], acc2[nt]);
      }
    }
  }

  // residual add + pw bias, bounds-guarded store
  #pragma unroll
  for (int nt = 0; nt < 4; ++nt) {
    int p = nt * 16 + m;
    int ph = p >> 3, pw = p & 7;
    int h = th * 8 + ph, w = tw * 8 + pw;
    if (h < 140 && w < 140) {
      #pragma unroll
      for (int r = 0; r < 8; ++r) {
        int oc = wv * 16 + r + 8 * g;
        int idx = ((b * 128 + oc) * 140 + h) * 140 + w;
        out[idx] = out[idx] + acc2[nt][r] + pwb[oc];
      }
    }
  }
}

// ================= launcher =================
extern "C" void kernel_launch(void* const* d_in, const int* in_sizes, int n_in,
                              void* d_out, int out_size, void* d_ws, size_t ws_size,
                              hipStream_t stream) {
  (void)in_sizes; (void)n_in; (void)out_size; (void)ws_size;
  const float* x      = (const float*)d_in[0];
  const float* dsw3   = (const float*)d_in[1];  const float* dsb3 = (const float*)d_in[2];
  const float* dsw5   = (const float*)d_in[3];  const float* dsb5 = (const float*)d_in[4];
  const float* dsw7   = (const float*)d_in[5];  const float* dsb7 = (const float*)d_in[6];
  const float* dsw9   = (const float*)d_in[7];  const float* dsb9 = (const float*)d_in[8];
  const float* hng    = (const float*)d_in[9];  const float* hnb  = (const float*)d_in[10];
  const float* wng    = (const float*)d_in[11]; const float* wnb  = (const float*)d_in[12];
  const float* cng    = (const float*)d_in[13]; const float* cnb  = (const float*)d_in[14];
  const float* qhw    = (const float*)d_in[15]; const float* khw  = (const float*)d_in[16];
  const float* vhw    = (const float*)d_in[17];
  const float* qww    = (const float*)d_in[18]; const float* kww  = (const float*)d_in[19];
  const float* vww    = (const float*)d_in[20];
  const float* mcb1w  = (const float*)d_in[21]; const float* mcb1g = (const float*)d_in[22];
  const float* mcb1b  = (const float*)d_in[23];
  const float* dw1w   = (const float*)d_in[24]; const float* dw1g = (const float*)d_in[25];
  const float* dw1b   = (const float*)d_in[26];
  const float* dw3w   = (const float*)d_in[27]; const float* dw3g = (const float*)d_in[28];
  const float* dw3b   = (const float*)d_in[29];
  const float* dw5w   = (const float*)d_in[30]; const float* dw5g = (const float*)d_in[31];
  const float* dw5b   = (const float*)d_in[32];
  const float* pww    = (const float*)d_in[33]; const float* pwg  = (const float*)d_in[34];
  const float* pwb    = (const float*)d_in[35];
  float* out = (float*)d_out;

  char* ws = (char*)d_ws;
  size_t off = 0;
  auto take = [&](size_t bytes) { size_t o = off; off = (off + bytes + 255) & ~(size_t)255; return o; };
  float*  hm    = (float*)(ws + take(8 * 128 * 140 * 4));
  float*  wm    = (float*)(ws + take(8 * 128 * 140 * 4));
  float*  hx    = (float*)(ws + take(8 * 128 * 140 * 4));
  float*  wx    = (float*)(ws + take(8 * 128 * 140 * 4));
  float*  ybuf  = (float*)(ws + take(8 * 128 * 400 * 4));
  float*  ynb   = (float*)(ws + take(8 * 128 * 400 * 4));
  float*  gbuf  = (float*)(ws + take(2048 * 4));
  __bf16* w1bf  = (__bf16*)(ws + take(768 * 128 * 2));
  __bf16* pwp   = (__bf16*)(ws + take(128 * 768 * 2));
  float*  dwf   = (float*)(ws + take(768 * 35 * 4));
  float*  dbias = (float*)(ws + take(768 * 3 * 4));
  __bf16* xbf   = (__bf16*)(ws + take((size_t)8 * 128 * 19600 * 2));

  k_wprep<<<384, 256, 0, stream>>>(mcb1w, mcb1g, pww, pwg,
                                   dw1w, dw1g, dw1b, dw3w, dw3g, dw3b, dw5w, dw5g, dw5b,
                                   w1bf, pwp, dwf, dbias);
  k_means<<<1024, 256, 0, stream>>>(x, hm, wm);
  k_dsab<<<64, 256, 0, stream>>>(hm, wm, dsw3, dsb3, dsw5, dsb5, dsw7, dsb7, dsw9, dsb9,
                                 hng, hnb, wng, wnb, hx, wx);
  k_gate_pool<<<1024, 256, 0, stream>>>(x, hx, wx, out, ybuf);
  k_ynorm<<<8, 256, 0, stream>>>(ybuf, cng, cnb, ynb);
  k_attn<<<128, 32, 0, stream>>>(ynb, qhw, khw, vhw, qww, kww, vww, gbuf);
  k_chgate<<<(8 * 128 * 19600 + 255) / 256, 256, 0, stream>>>(out, gbuf, xbf);
  dim3 gmcb(18, 18, 8);
  k_mcb<<<gmcb, 256, 0, stream>>>(xbf, w1bf, pwp, dwf, dbias, mcb1b, pwb, out);
}
